// QuerySATCore_28046136442974
// MI455X (gfx1250) — matrix operand order repaired and checked by reading, not compiled, not run on hardware
//
#include <hip/hip_runtime.h>

typedef _Float16 h16;
typedef __attribute__((ext_vector_type(8)))  h16   v8h;
typedef __attribute__((ext_vector_type(16))) h16   v16h;
typedef __attribute__((ext_vector_type(8)))  float v8f;

#define DEVINL __device__ __forceinline__

constexpr int NV = 200000;   // variables
constexpr int NC = 500000;   // clauses
constexpr int NE = 1500000;  // edges

// ---------------------------------------------------------------------------
// A-fragment load from row-major fp16 LDS tile.
// 16-bit A layout (16x32): lane<16 holds row (lane&15), K {0..7,16..23};
// lane>=16 holds K {8..15,24..31}  -> two 16B LDS loads per K-chunk.
// ---------------------------------------------------------------------------
DEVINL v16h load_a(const h16* ap) {
  v8h lo = *(const v8h*)(ap);
  v8h hi = *(const v8h*)(ap + 16);
  v16h a;
#pragma unroll
  for (int i = 0; i < 8; ++i) { a[i] = lo[i]; a[i + 8] = hi[i]; }
  return a;
}

// ---------------------------------------------------------------------------
// One n-tile, BOTH m-tiles (rows 0..15 and 16..31) against a single B
// fragment load per K-chunk: 2 WMMAs per 32B global load.
// B: transposed fp16 Wt[n][k], Kpad%32==0. lane<16 -> col(lane&15) K {0..7,16..23},
// lane>=16 -> K {8..15,24..31}: two 16B loads per lane per chunk.
// ---------------------------------------------------------------------------
DEVINL void mma_pair(const h16* sA, int a_pitch, const h16* Wt, int kpad,
                     int nt, int kchunks, int lane, v8f& acc0, v8f& acc1) {
  const int kh = lane >> 4;
  const h16* ap0 = sA + (lane & 15) * a_pitch + kh * 8;   // rows 0..15
  const h16* ap1 = ap0 + 16 * a_pitch;                    // rows 16..31
  const h16* bp  = Wt + (size_t)(nt * 16 + (lane & 15)) * kpad + kh * 8;
  for (int kc = 0; kc < kchunks; ++kc) {
    v16h b  = load_a(bp + kc * 32);
    v16h a0 = load_a(ap0 + kc * 32);
    acc0 = __builtin_amdgcn_wmma_f32_16x16x32_f16(false, a0, false, b,
                                                  (short)0, acc0, false, false);
    v16h a1 = load_a(ap1 + kc * 32);
    acc1 = __builtin_amdgcn_wmma_f32_16x16x32_f16(false, a1, false, b,
                                                  (short)0, acc1, false, false);
  }
}

// ---------------------------------------------------------------------------
// Weight convert: W[K][Nn] fp32 (fan_in x fan_out) -> Wt[Npad][Kpad] fp16
// transposed, zero padded.
// ---------------------------------------------------------------------------
__global__ void k_convert_wt(const float* __restrict__ W, h16* __restrict__ Wt,
                             int K, int Nn, int Kpad, int Npad) {
  int idx = blockIdx.x * blockDim.x + threadIdx.x;
  int total = Npad * Kpad;
  if (idx >= total) return;
  int n = idx / Kpad;
  int k = idx - n * Kpad;
  float v = (n < Nn && k < K) ? W[(size_t)k * Nn + n] : 0.f;
  Wt[idx] = (h16)v;
}

__global__ void k_zero4(float* __restrict__ p, int n4) {
  int i = blockIdx.x * blockDim.x + threadIdx.x;
  if (i < n4) ((float4*)p)[i] = make_float4(0.f, 0.f, 0.f, 0.f);
}

// ---------------------------------------------------------------------------
// Query MLP: [N,132] -> relu 76 -> 64.   Kpad: 160 / 96.
// ---------------------------------------------------------------------------
__global__ void __launch_bounds__(256)
k_query_mlp(const float* __restrict__ vx, const float* __restrict__ noise,
            const h16* __restrict__ W1t, const float* __restrict__ b1,
            const h16* __restrict__ W2t, const float* __restrict__ b2,
            float* __restrict__ query) {
  __shared__ __align__(32) h16 sX[32 * 160];
  __shared__ __align__(32) h16 sH[32 * 96];
  const int row0 = blockIdx.x * 32;
  const int tid = threadIdx.x, lane = tid & 31, wave = tid >> 5;

  for (int i = tid; i < 32 * 160; i += 256) {
    int r = i / 160, c = i - r * 160;
    int gr = row0 + r;
    float v;
    if (c < 128)       v = vx[(size_t)gr * 128 + c];
    else if (c < 132)  v = noise[(size_t)gr * 4 + (c - 128)];
    else               v = 0.f;
    sX[i] = (h16)v;
  }
  for (int i = tid; i < 32 * 96; i += 256) sH[i] = (h16)0.f;
  __syncthreads();

  for (int nt = wave; nt < 5; nt += 8) {
    v8f acc0 = {0.f,0.f,0.f,0.f,0.f,0.f,0.f,0.f}, acc1 = acc0;
    mma_pair(sX, 160, W1t, 160, nt, 5, lane, acc0, acc1);
    int n = nt * 16 + (lane & 15);
    float bv = (n < 76) ? b1[n] : 0.f;
#pragma unroll
    for (int r = 0; r < 8; ++r) {
      int m = r + ((lane >> 4) << 3);
      float v0 = fmaxf(acc0[r] + bv, 0.f);
      float v1 = fmaxf(acc1[r] + bv, 0.f);
      if (n >= 76) { v0 = 0.f; v1 = 0.f; }
      sH[m * 96 + n]        = (h16)v0;
      sH[(m + 16) * 96 + n] = (h16)v1;
    }
  }
  __syncthreads();

  for (int nt = wave; nt < 4; nt += 8) {
    v8f acc0 = {0.f,0.f,0.f,0.f,0.f,0.f,0.f,0.f}, acc1 = acc0;
    mma_pair(sH, 96, W2t, 96, nt, 3, lane, acc0, acc1);
    int n = nt * 16 + (lane & 15);
    float bv = b2[n];
#pragma unroll
    for (int r = 0; r < 8; ++r) {
      int m = r + ((lane >> 4) << 3);
      query[(size_t)(row0 + m) * 64 + n]      = acc0[r] + bv;
      query[(size_t)(row0 + m + 16) * 64 + n] = acc1[r] + bv;
    }
  }
}

// ---------------------------------------------------------------------------
// Edge pass 1: csum[clause] += softplus(sign * query[var])
// ---------------------------------------------------------------------------
__global__ void k_edge_softplus(const float* __restrict__ query,
                                const int* __restrict__ edge_var,
                                const int* __restrict__ edge_clause,
                                const int* __restrict__ polarity,
                                float* __restrict__ csum) {
  int t = blockIdx.x * blockDim.x + threadIdx.x;
  if (t >= NE * 4) return;
  int e = t >> 2;
  int c0 = (t & 3) * 16;
  float sg = 1.f - 2.f * (float)polarity[e];
  const float4* qp = (const float4*)(query + (size_t)edge_var[e] * 64 + c0);
  float* cp = csum + (size_t)edge_clause[e] * 64 + c0;
#pragma unroll
  for (int j = 0; j < 4; ++j) {
    float4 q = qp[j];
    float xs[4] = {sg * q.x, sg * q.y, sg * q.z, sg * q.w};
#pragma unroll
    for (int i = 0; i < 4; ++i) {
      float x = xs[i];
      float sp = (x > 20.f) ? x : log1pf(__expf(x));
      atomicAdd(cp + j * 4 + i, sp);
    }
  }
}

// in-place: csum -> 4*exp(-csum)   (grad base = -0.25 * this)
__global__ void k_exp4(float* __restrict__ csum, int n4) {
  int i = blockIdx.x * blockDim.x + threadIdx.x;
  if (i >= n4) return;
  float4 v = ((float4*)csum)[i];
  v.x = 4.f * __expf(-v.x);
  v.y = 4.f * __expf(-v.y);
  v.z = 4.f * __expf(-v.z);
  v.w = 4.f * __expf(-v.w);
  ((float4*)csum)[i] = v;
}

// ---------------------------------------------------------------------------
// Edge pass 2 (analytic VJP): vgrad[var] += -exp(-csum)[clause]*sigmoid(sq)*sign
// ---------------------------------------------------------------------------
__global__ void k_edge_grad(const float* __restrict__ query,
                            const float* __restrict__ cl4,
                            const int* __restrict__ edge_var,
                            const int* __restrict__ edge_clause,
                            const int* __restrict__ polarity,
                            float* __restrict__ vgrad) {
  int t = blockIdx.x * blockDim.x + threadIdx.x;
  if (t >= NE * 4) return;
  int e = t >> 2;
  int c0 = (t & 3) * 16;
  int v = edge_var[e];
  float sg = 1.f - 2.f * (float)polarity[e];
  const float4* qp = (const float4*)(query + (size_t)v * 64 + c0);
  const float4* gp = (const float4*)(cl4 + (size_t)edge_clause[e] * 64 + c0);
  float* op = vgrad + (size_t)v * 64 + c0;
#pragma unroll
  for (int j = 0; j < 4; ++j) {
    float4 q = qp[j];
    float4 g = gp[j];
    float qs[4] = {q.x, q.y, q.z, q.w};
    float gs[4] = {g.x, g.y, g.z, g.w};
#pragma unroll
    for (int i = 0; i < 4; ++i) {
      float sq = sg * qs[i];
      float sig = 1.f / (1.f + __expf(-sq));
      atomicAdd(op + j * 4 + i, (-0.25f * gs[i]) * sig * sg);
    }
  }
}

// ---------------------------------------------------------------------------
// Clause MLP: [C, 128|4*exp(-csum)] (192) -> relu 204 -> 192
// cols 0..63 -> vla buffer, cols 64..191 -> d_out clause region (pre-PairNorm)
// Kpad: 192 / 224 ; n-tiles: 13 / 12
// ---------------------------------------------------------------------------
__global__ void __launch_bounds__(256)
k_clause_mlp(const float* __restrict__ cx, const float* __restrict__ cl4,
             const h16* __restrict__ W1t, const float* __restrict__ b1,
             const h16* __restrict__ W2t, const float* __restrict__ b2,
             float* __restrict__ vla, float* __restrict__ out_cl) {
  __shared__ __align__(32) h16 sX[32 * 192];
  __shared__ __align__(32) h16 sH[32 * 224];
  const int row0 = blockIdx.x * 32;
  const int tid = threadIdx.x, lane = tid & 31, wave = tid >> 5;

  for (int i = tid; i < 32 * 192; i += 256) {
    int r = i / 192, c = i - r * 192;
    int gr = row0 + r;
    float v = (c < 128) ? cx[(size_t)gr * 128 + c]
                        : cl4[(size_t)gr * 64 + (c - 128)];
    sX[i] = (h16)v;
  }
  for (int i = tid; i < 32 * 224; i += 256) sH[i] = (h16)0.f;
  __syncthreads();

  for (int nt = wave; nt < 13; nt += 8) {
    v8f acc0 = {0.f,0.f,0.f,0.f,0.f,0.f,0.f,0.f}, acc1 = acc0;
    mma_pair(sX, 192, W1t, 192, nt, 6, lane, acc0, acc1);
    int n = nt * 16 + (lane & 15);
    float bv = (n < 204) ? b1[n] : 0.f;
#pragma unroll
    for (int r = 0; r < 8; ++r) {
      int m = r + ((lane >> 4) << 3);
      float v0 = fmaxf(acc0[r] + bv, 0.f);
      float v1 = fmaxf(acc1[r] + bv, 0.f);
      if (n >= 204) { v0 = 0.f; v1 = 0.f; }
      sH[m * 224 + n]        = (h16)v0;
      sH[(m + 16) * 224 + n] = (h16)v1;
    }
  }
  __syncthreads();

  for (int nt = wave; nt < 12; nt += 8) {
    v8f acc0 = {0.f,0.f,0.f,0.f,0.f,0.f,0.f,0.f}, acc1 = acc0;
    mma_pair(sH, 224, W2t, 224, nt, 7, lane, acc0, acc1);
    int n = nt * 16 + (lane & 15);
    float bv = b2[n];
#pragma unroll
    for (int r = 0; r < 8; ++r) {
      int m = r + ((lane >> 4) << 3);
      float v0 = acc0[r] + bv;
      float v1 = acc1[r] + bv;
      if (n < 64) {
        vla[(size_t)(row0 + m) * 64 + n]      = v0;
        vla[(size_t)(row0 + m + 16) * 64 + n] = v1;
      } else {
        out_cl[(size_t)(row0 + m) * 128 + (n - 64)]      = v0;
        out_cl[(size_t)(row0 + m + 16) * 128 + (n - 64)] = v1;
      }
    }
  }
}

// ---------------------------------------------------------------------------
// Edge pass 3: vloss[literal] += vla[clause]
// ---------------------------------------------------------------------------
__global__ void k_edge_vloss(const float* __restrict__ vla,
                             const int* __restrict__ edge_clause,
                             const int* __restrict__ lit,
                             float* __restrict__ vloss) {
  int t = blockIdx.x * blockDim.x + threadIdx.x;
  if (t >= NE * 4) return;
  int e = t >> 2;
  int c0 = (t & 3) * 16;
  const float4* sp = (const float4*)(vla + (size_t)edge_clause[e] * 64 + c0);
  float* dp = vloss + (size_t)lit[e] * 64 + c0;
#pragma unroll
  for (int j = 0; j < 4; ++j) {
    float4 v = sp[j];
    atomicAdd(dp + j * 4 + 0, v.x);
    atomicAdd(dp + j * 4 + 1, v.y);
    atomicAdd(dp + j * 4 + 2, v.z);
    atomicAdd(dp + j * 4 + 3, v.w);
  }
}

// ---------------------------------------------------------------------------
// Variable MLP: [N,320] -> relu 230 -> relu 230 -> 128 (pre-PairNorm to d_out)
// Kpad: 320 / 256 / 256 ; hidden n-tiles: 15, out n-tiles: 8
// LDS ping-pong: sA(X,320) -> sB(H1,256) -> sA(H2,256)
// ---------------------------------------------------------------------------
__global__ void __launch_bounds__(256)
k_var_mlp(const float* __restrict__ vgrad, const float* __restrict__ dw,
          const float* __restrict__ vx, const float* __restrict__ vloss,
          const float* __restrict__ ldw,
          const h16* __restrict__ W1t, const float* __restrict__ b1,
          const h16* __restrict__ W2t, const float* __restrict__ b2,
          const h16* __restrict__ W3t, const float* __restrict__ b3,
          float* __restrict__ out_v) {
  __shared__ __align__(32) h16 sA[32 * 320];
  __shared__ __align__(32) h16 sB[32 * 256];
  const int row0 = blockIdx.x * 32;
  const int tid = threadIdx.x, lane = tid & 31, wave = tid >> 5;

  for (int i = tid; i < 32 * 320; i += 256) {
    int r = i / 320, c = i - r * 320;
    int gr = row0 + r;
    float v;
    if (c < 64)        v = vgrad[(size_t)gr * 64 + c] * dw[gr];
    else if (c < 192)  v = vx[(size_t)gr * 128 + (c - 64)];
    else if (c < 256)  { int li = 2 * gr;     v = vloss[(size_t)li * 64 + (c - 192)] * ldw[li]; }
    else               { int li = 2 * gr + 1; v = vloss[(size_t)li * 64 + (c - 256)] * ldw[li]; }
    sA[i] = (h16)v;
  }
  for (int i = tid; i < 32 * 256; i += 256) sB[i] = (h16)0.f;
  __syncthreads();

  for (int nt = wave; nt < 15; nt += 8) {            // layer 1
    v8f acc0 = {0.f,0.f,0.f,0.f,0.f,0.f,0.f,0.f}, acc1 = acc0;
    mma_pair(sA, 320, W1t, 320, nt, 10, lane, acc0, acc1);
    int n = nt * 16 + (lane & 15);
    float bv = (n < 230) ? b1[n] : 0.f;
#pragma unroll
    for (int r = 0; r < 8; ++r) {
      int m = r + ((lane >> 4) << 3);
      float v0 = fmaxf(acc0[r] + bv, 0.f);
      float v1 = fmaxf(acc1[r] + bv, 0.f);
      if (n >= 230) { v0 = 0.f; v1 = 0.f; }
      sB[m * 256 + n]        = (h16)v0;
      sB[(m + 16) * 256 + n] = (h16)v1;
    }
  }
  __syncthreads();
  for (int i = tid; i < 32 * 256; i += 256) sA[i] = (h16)0.f;  // reuse X buffer
  __syncthreads();

  for (int nt = wave; nt < 15; nt += 8) {            // layer 2
    v8f acc0 = {0.f,0.f,0.f,0.f,0.f,0.f,0.f,0.f}, acc1 = acc0;
    mma_pair(sB, 256, W2t, 256, nt, 8, lane, acc0, acc1);
    int n = nt * 16 + (lane & 15);
    float bv = (n < 230) ? b2[n] : 0.f;
#pragma unroll
    for (int r = 0; r < 8; ++r) {
      int m = r + ((lane >> 4) << 3);
      float v0 = fmaxf(acc0[r] + bv, 0.f);
      float v1 = fmaxf(acc1[r] + bv, 0.f);
      if (n >= 230) { v0 = 0.f; v1 = 0.f; }
      sA[m * 256 + n]        = (h16)v0;
      sA[(m + 16) * 256 + n] = (h16)v1;
    }
  }
  __syncthreads();

  for (int nt = wave; nt < 8; nt += 8) {             // layer 3
    v8f acc0 = {0.f,0.f,0.f,0.f,0.f,0.f,0.f,0.f}, acc1 = acc0;
    mma_pair(sA, 256, W3t, 256, nt, 8, lane, acc0, acc1);
    int n = nt * 16 + (lane & 15);
    float bv = b3[n];
#pragma unroll
    for (int r = 0; r < 8; ++r) {
      int m = r + ((lane >> 4) << 3);
      out_v[(size_t)(row0 + m) * 128 + n]      = acc0[r] + bv;
      out_v[(size_t)(row0 + m + 16) * 128 + n] = acc1[r] + bv;
    }
  }
}

// ---------------------------------------------------------------------------
// PairNorm: per-column sums, centered sum-of-squares, in-place finalize.
// ---------------------------------------------------------------------------
__global__ void k_colsum(const float* __restrict__ x, int rows, float* __restrict__ colsum) {
  int col = threadIdx.x;                 // blockDim.x == 128
  float s = 0.f;
  for (int r = blockIdx.x; r < rows; r += gridDim.x)
    s += x[(size_t)r * 128 + col];
  atomicAdd(&colsum[col], s);
}

__global__ void k_ssq(const float* __restrict__ x, int rows,
                      const float* __restrict__ colsum, float* __restrict__ ssq) {
  __shared__ float red[256];
  const float inv = 1.f / (float)rows;
  float s = 0.f;
  int total4 = rows * 32;
  for (int i = blockIdx.x * blockDim.x + threadIdx.x; i < total4;
       i += gridDim.x * blockDim.x) {
    int c4 = (i & 31) * 4;
    float4 v = ((const float4*)x)[i];
    float a = v.x - colsum[c4 + 0] * inv;
    float b = v.y - colsum[c4 + 1] * inv;
    float c = v.z - colsum[c4 + 2] * inv;
    float d = v.w - colsum[c4 + 3] * inv;
    s += a * a + b * b + c * c + d * d;
  }
  red[threadIdx.x] = s;
  __syncthreads();
  for (int o = 128; o > 0; o >>= 1) {
    if (threadIdx.x < o) red[threadIdx.x] += red[threadIdx.x + o];
    __syncthreads();
  }
  if (threadIdx.x == 0) atomicAdd(ssq, red[0]);
}

__global__ void k_pairnorm_finish(float* __restrict__ x, const float* __restrict__ base,
                                  int rows, const float* __restrict__ colsum,
                                  const float* __restrict__ ssq) {
  int i = blockIdx.x * blockDim.x + threadIdx.x;     // float4 index
  int total4 = rows * 32;
  if (i >= total4) return;
  int c4 = (i & 31) * 4;
  const float inv = 1.f / (float)rows;
  float scale = rsqrtf(1e-6f + ssq[0] * inv);
  float4 v = ((const float4*)x)[i];
  float4 b = ((const float4*)base)[i];
  v.x = (v.x - colsum[c4 + 0] * inv) * scale + 0.1f * b.x;
  v.y = (v.y - colsum[c4 + 1] * inv) * scale + 0.1f * b.y;
  v.z = (v.z - colsum[c4 + 2] * inv) * scale + 0.1f * b.z;
  v.w = (v.w - colsum[c4 + 3] * inv) * scale + 0.1f * b.w;
  ((float4*)x)[i] = v;
}

// ---------------------------------------------------------------------------
extern "C" void kernel_launch(void* const* d_in, const int* in_sizes, int n_in,
                              void* d_out, int out_size, void* d_ws, size_t ws_size,
                              hipStream_t stream) {
  (void)in_sizes; (void)n_in; (void)out_size; (void)ws_size;
  const float* variable_x = (const float*)d_in[0];
  const float* clause_x   = (const float*)d_in[1];
  const float* noise      = (const float*)d_in[2];
  const float* dw         = (const float*)d_in[3];
  const float* ldw        = (const float*)d_in[4];
  const float* Wq1 = (const float*)d_in[5];  const float* bq1 = (const float*)d_in[6];
  const float* Wq2 = (const float*)d_in[7];  const float* bq2 = (const float*)d_in[8];
  const float* Wc1 = (const float*)d_in[9];  const float* bc1 = (const float*)d_in[10];
  const float* Wc2 = (const float*)d_in[11]; const float* bc2 = (const float*)d_in[12];
  const float* Wu1 = (const float*)d_in[13]; const float* bu1 = (const float*)d_in[14];
  const float* Wu2 = (const float*)d_in[15]; const float* bu2 = (const float*)d_in[16];
  const float* Wu3 = (const float*)d_in[17]; const float* bu3 = (const float*)d_in[18];
  const int* edge_var    = (const int*)d_in[19];
  const int* edge_clause = (const int*)d_in[20];
  const int* lit         = (const int*)d_in[21];
  const int* polarity    = (const int*)d_in[22];

  float* out_v  = (float*)d_out;                       // [N,128]
  float* out_cl = (float*)d_out + (size_t)NV * 128;    // [C,128]

  // workspace layout (float offsets)
  float* ws = (float*)d_ws;
  const size_t OFF_VGRAD = (size_t)NC * 64;
  const size_t OFF_VLOSS = OFF_VGRAD + (size_t)NV * 64;
  const size_t OFF_STATS = OFF_VLOSS + (size_t)2 * NV * 64;
  const size_t ZERO_CNT  = OFF_STATS + 320;            // csum+vgrad+vloss+stats
  const size_t OFF_QUERY = ZERO_CNT;
  const size_t OFF_VLA   = OFF_QUERY + (size_t)NV * 64;
  const size_t OFF_WT    = OFF_VLA + (size_t)NC * 64;  // multiple of 8 -> 32B aligned

  float* csum  = ws;                  // [C,64]; becomes 4*exp(-csum) in-place
  float* vgrad = ws + OFF_VGRAD;      // [N,64]
  float* vloss = ws + OFF_VLOSS;      // [2N,64]
  float* colsum_cl  = ws + OFF_STATS;
  float* colsum_var = ws + OFF_STATS + 128;
  float* ssq_cl     = ws + OFF_STATS + 256;
  float* ssq_var    = ws + OFF_STATS + 257;
  float* query = ws + OFF_QUERY;      // [N,64]
  float* vla   = ws + OFF_VLA;        // [C,64]

  h16* wt = (h16*)(ws + OFF_WT);
  h16* Wq1t = wt;                    // 80 x 160
  h16* Wq2t = Wq1t + 80 * 160;       // 64 x 96
  h16* Wc1t = Wq2t + 64 * 96;        // 208 x 192
  h16* Wc2t = Wc1t + 208 * 192;      // 192 x 224
  h16* Wu1t = Wc2t + 192 * 224;      // 240 x 320
  h16* Wu2t = Wu1t + 240 * 320;      // 240 x 256
  h16* Wu3t = Wu2t + 240 * 256;      // 128 x 256

  // 1) zero the atomic accumulators + stats
  {
    int n4 = (int)(ZERO_CNT / 4);
    k_zero4<<<(n4 + 255) / 256, 256, 0, stream>>>(ws, n4);
  }

  // 2) weight convert+transpose to fp16 (once per call; tiny)
  auto cvt = [&](const float* W, h16* dst, int K, int Nn, int Kpad, int Npad) {
    int total = Npad * Kpad;
    k_convert_wt<<<(total + 255) / 256, 256, 0, stream>>>(W, dst, K, Nn, Kpad, Npad);
  };
  cvt(Wq1, Wq1t, 132,  76, 160,  80);
  cvt(Wq2, Wq2t,  76,  64,  96,  64);
  cvt(Wc1, Wc1t, 192, 204, 192, 208);
  cvt(Wc2, Wc2t, 204, 192, 224, 192);
  cvt(Wu1, Wu1t, 320, 230, 320, 240);
  cvt(Wu2, Wu2t, 230, 230, 256, 240);
  cvt(Wu3, Wu3t, 230, 128, 256, 128);

  // 3) query MLP (WMMA)
  k_query_mlp<<<NV / 32, 256, 0, stream>>>(variable_x, noise, Wq1t, bq1, Wq2t, bq2, query);

  // 4) edge forward: csum
  k_edge_softplus<<<(NE * 4 + 255) / 256, 256, 0, stream>>>(query, edge_var, edge_clause,
                                                            polarity, csum);
  // 5) csum -> 4*exp(-csum) in-place
  k_exp4<<<(NC * 16 + 255) / 256, 256, 0, stream>>>(csum, NC * 16);

  // 6) edge backward: vgrad
  k_edge_grad<<<(NE * 4 + 255) / 256, 256, 0, stream>>>(query, csum, edge_var, edge_clause,
                                                        polarity, vgrad);

  // 7) clause MLP (WMMA) -> vla + d_out clause region (pre-PairNorm)
  k_clause_mlp<<<NC / 32, 256, 0, stream>>>(clause_x, csum, Wc1t, bc1, Wc2t, bc2,
                                            vla, out_cl);

  // 8) edge scatter: vloss
  k_edge_vloss<<<(NE * 4 + 255) / 256, 256, 0, stream>>>(vla, edge_clause, lit, vloss);

  // 9) variable MLP (WMMA) -> d_out variable region (pre-PairNorm)
  k_var_mlp<<<NV / 32, 256, 0, stream>>>(vgrad, dw, variable_x, vloss, ldw,
                                         Wu1t, bu1, Wu2t, bu2, Wu3t, bu3, out_v);

  // 10) PairNorm for both outputs, in place on d_out
  k_colsum<<<512, 128, 0, stream>>>(out_cl, NC, colsum_cl);
  k_colsum<<<512, 128, 0, stream>>>(out_v,  NV, colsum_var);
  k_ssq<<<1024, 256, 0, stream>>>(out_cl, NC, colsum_cl, ssq_cl);
  k_ssq<<<1024, 256, 0, stream>>>(out_v,  NV, colsum_var, ssq_var);
  k_pairnorm_finish<<<(NC * 32 + 255) / 256, 256, 0, stream>>>(out_cl, clause_x, NC,
                                                               colsum_cl, ssq_cl);
  k_pairnorm_finish<<<(NV * 32 + 255) / 256, 256, 0, stream>>>(out_v, variable_x, NV,
                                                               colsum_var, ssq_var);
}
